// dLSTM_884763263339
// MI455X (gfx1250) — compile-verified, matrix operand order checked by reading, not run on hardware
//
#include <hip/hip_runtime.h>
#include <hip/hip_bf16.h>

// Problem constants (match reference)
constexpr int LL = 2;      // layers
constexpr int BB = 64;     // batch (GEMM M)
constexpr int HH = 1024;   // hidden
constexpr int TT = 512;    // seq_len
constexpr int GRIDN = HH / 32;              // 32 persistent workgroups
constexpr int BH  = BB * HH;
constexpr int LBH = LL * BB * HH;

typedef __attribute__((ext_vector_type(16))) __bf16 v16bf;
typedef __attribute__((ext_vector_type(8)))  float  v8f;

union ABpack { v16bf v; uint4 q[2]; };

__device__ __forceinline__ unsigned short f2bf(float f) {
    unsigned u = __float_as_uint(f);
    unsigned r = u + 0x7fffu + ((u >> 16) & 1u);   // round-to-nearest-even
    return (unsigned short)(r >> 16);
}

__device__ __forceinline__ float sigm(float x) {
    return 1.0f / (1.0f + __expf(-x));
}

__device__ __forceinline__ void load_a(ABpack& a, const unsigned short* arow,
                                       int kbase, int half) {
    a.q[0] = *(const uint4*)(arow + kbase + half * 8);        // K = kb+half*8 .. +7
    a.q[1] = *(const uint4*)(arow + kbase + 16 + half * 8);   // K = kb+16+half*8 ..
}
__device__ __forceinline__ void load_b(ABpack& b, const unsigned short* bp) {
    b.q[0] = *(const uint4*)(bp);
    b.q[1] = *(const uint4*)(bp + 8);
}

// Device-wide barrier: monotonic counter, release-add + acquire-spin at agent
// scope (emits the L2 wb/inv needed for cross-WG h/x handoff).
__device__ __forceinline__ void grid_barrier(unsigned* counter, unsigned target) {
    __syncthreads();
    if (threadIdx.x == 0) {
        __hip_atomic_fetch_add(counter, 1u, __ATOMIC_RELEASE, __HIP_MEMORY_SCOPE_AGENT);
        while (__hip_atomic_load(counter, __ATOMIC_ACQUIRE, __HIP_MEMORY_SCOPE_AGENT) < target)
            __builtin_amdgcn_s_sleep(1);
    }
    __syncthreads();
}

// ---------------------------------------------------------------------------
// Pack fp32 weight matrix W[rows, K] (used as B = W^T) into bf16 in the
// per-lane order of the 32x16 bf16 WMMA B operand:
//   dest d = ((jt*(K/32) + kt)*32 + lane)*16 + e
//   n = jt*16 + lane%16 ; k = kt*32 + (e/8)*16 + (lane/16)*8 + (e%8)
// ---------------------------------------------------------------------------
__global__ void pack_w_kernel(const float* __restrict__ W,
                              unsigned short* __restrict__ P,
                              int K, long total) {
    long d = (long)blockIdx.x * blockDim.x + threadIdx.x;
    if (d >= total) return;
    int e    = (int)(d & 15);
    int lane = (int)((d >> 4) & 31);
    long rest = d >> 9;
    int ktiles = K >> 5;
    int kt = (int)(rest % ktiles);
    long jt = rest / ktiles;
    int  n  = (int)(jt << 4) + (lane & 15);
    int  k  = (kt << 5) + ((e >> 3) << 4) + ((lane >> 4) << 3) + (e & 7);
    P[d] = f2bf(W[(long)n * K + k]);
}

// ---------------------------------------------------------------------------
// Init: h0 -> bf16 ping buffer, c0 -> fp32 staging, combined bias, x0 = 0,
// and re-zero the grid-barrier counter (required for graph replay).
// ---------------------------------------------------------------------------
__global__ void lstm_init_kernel(const float* __restrict__ h0,
                                 const float* __restrict__ c0,
                                 const float* __restrict__ b_ih,
                                 const float* __restrict__ b_hh,
                                 unsigned short* __restrict__ hbuf0,
                                 float* __restrict__ cst,
                                 float* __restrict__ bias,
                                 unsigned short* __restrict__ xbf,
                                 unsigned* __restrict__ counter) {
    int i = blockIdx.x * blockDim.x + threadIdx.x;
    if (i == 0) *counter = 0u;
    if (i < LBH)         { hbuf0[i] = f2bf(h0[i]); cst[i] = c0[i]; }
    if (i < LL * 4 * HH) { bias[i] = b_ih[i] + b_hh[i]; }
    if (i < BH)          { xbf[i] = 0; }
}

// ---------------------------------------------------------------------------
// Split-K 4-gate GEMM core: this wave accumulates its source's contribution
// to the four 16x16 gate tiles at column block n0 (pre-swizzled bf16 B).
// ---------------------------------------------------------------------------
__device__ __forceinline__ void gate_gemm(v8f acc[4],
                                          const unsigned short* __restrict__ arow,
                                          const unsigned short* __restrict__ W,
                                          int n0, int lane, int half) {
    constexpr int KT = HH / 32;
    const unsigned short* bbase[4];
    #pragma unroll
    for (int q = 0; q < 4; ++q)
        bbase[q] = W + (long)((q * HH + n0) >> 4) * KT * 512 + lane * 16;
    for (int kt = 0; kt < KT; ++kt) {
        ABpack a;
        load_a(a, arow, kt * 32, half);
        #pragma unroll
        for (int q = 0; q < 4; ++q) {
            ABpack b;
            load_b(b, bbase[q] + kt * 512);
            acc[q] = __builtin_amdgcn_wmma_f32_16x16x32_bf16(
                false, a.v, false, b.v, (short)0, acc[q], false, false);
        }
    }
}

// ---------------------------------------------------------------------------
// Persistent LSTM: the ENTIRE 512-step recurrence in one kernel.
// Grid: 32 WGs x 512 threads (16 waves: mstrip 0..3, colg 0..1, khalf 0..1).
// Each WG owns a 32-column slice of H for both layers; c-state for both
// layers lives in registers for all 512 steps. Per step:
//   phase0: layer-0 gates GEMM (khalf splits x- vs h-source) -> LDS reduce ->
//           fused pointwise -> bf16 h (ping-pong)            -> grid barrier
//   phase1: layer-1 likewise, fp32 h streamed to d_out       -> grid barrier
//   phase2: projection x_next = h1 @ out_w^T + out_b -> xbf  -> grid barrier
// Grid barriers replace ~1536 kernel launches.
// ---------------------------------------------------------------------------
__global__ __launch_bounds__(512) void lstm_persistent(
    const unsigned short* __restrict__ pk_wih,   // packed bf16 [L][4H x H]
    const unsigned short* __restrict__ pk_whh,   // packed bf16 [L][4H x H]
    const unsigned short* __restrict__ pk_outw,  // packed bf16 [H x H]
    const float* __restrict__ bias,              // [L,4H] (b_ih + b_hh)
    const float* __restrict__ out_b,             // [H]
    const float* __restrict__ cst,               // [L,B,H] initial c
    unsigned short* __restrict__ hbuf,           // [2][L,B,H] bf16 ping-pong
    unsigned short* __restrict__ xbf,            // [B,H] bf16 feedback input
    float* __restrict__ out,                     // [T,B,H]
    unsigned* __restrict__ counter)              // grid-barrier counter
{
    __shared__ float red[8][4][8][32];           // [mstrip*2+colg][gate][r][lane] = 32 KB

    const int lane   = threadIdx.x & 31;
    const int wave   = threadIdx.x >> 5;
    const int mstrip = wave & 3;
    const int colg   = (wave >> 2) & 1;
    const int khalf  = wave >> 3;                // 0: input-source, 1: h-source
    const int n0     = blockIdx.x * 32 + colg * 16;
    const int m      = mstrip * 16 + (lane & 15);   // A-operand row
    const int half   = lane >> 4;                   // A-operand K-half
    const int n      = n0 + (lane & 15);            // C-operand column
    const int tile   = mstrip * 2 + colg;
    constexpr int KT = HH / 32;
    constexpr size_t W4 = (size_t)4 * HH * HH;      // per-layer packed weight stride

    // Register-resident cell state + biases (pointwise waves only).
    float creg[LL][8];
    float bz[LL][4];
    float obias = 0.0f;
    if (khalf == 0) {
        #pragma unroll
        for (int l = 0; l < LL; ++l) {
            #pragma unroll
            for (int r = 0; r < 8; ++r) {
                const int mrow = mstrip * 16 + half * 8 + r;
                creg[l][r] = cst[(size_t)l * BH + (size_t)mrow * HH + n];
            }
            #pragma unroll
            for (int g = 0; g < 4; ++g)
                bz[l][g] = bias[l * 4 * HH + g * HH + n];
        }
        obias = out_b[n];
    }

    unsigned done = 0;
    int p = 0;
    for (int t = 0; t < TT; ++t) {
        // ---- phases 0,1: the two LSTM layers ----
        #pragma unroll 1
        for (int l = 0; l < LL; ++l) {
            const unsigned short* Ain  = (l == 0) ? xbf
                                                  : hbuf + (size_t)(1 - p) * LBH;
            const unsigned short* Ah   = hbuf + (size_t)p * LBH + (size_t)l * BH;
            const unsigned short* Wsrc = (khalf ? pk_whh : pk_wih) + (size_t)l * W4;
            const unsigned short* arow = (khalf ? Ah : Ain) + (size_t)m * HH;

            v8f acc[4] = {};
            gate_gemm(acc, arow, Wsrc, n0, lane, half);

            if (khalf == 1) {
                #pragma unroll
                for (int q = 0; q < 4; ++q)
                    #pragma unroll
                    for (int r = 0; r < 8; ++r)
                        red[tile][q][r][lane] = acc[q][r];
            }
            __syncthreads();
            if (khalf == 0) {
                unsigned short* hnew = hbuf + (size_t)(1 - p) * LBH + (size_t)l * BH;
                float* fout = (l == LL - 1) ? out + (size_t)t * BH : nullptr;
                #pragma unroll
                for (int r = 0; r < 8; ++r) {
                    const int mrow = mstrip * 16 + half * 8 + r;
                    const size_t idx = (size_t)mrow * HH + n;
                    const float gi = acc[0][r] + red[tile][0][r][lane] + bz[l][0];
                    const float gf = acc[1][r] + red[tile][1][r][lane] + bz[l][1];
                    const float gg = acc[2][r] + red[tile][2][r][lane] + bz[l][2];
                    const float go = acc[3][r] + red[tile][3][r][lane] + bz[l][3];
                    const float cn = sigm(gf) * creg[l][r] + sigm(gi) * tanhf(gg);
                    creg[l][r] = cn;
                    const float hn = sigm(go) * tanhf(cn);
                    hnew[idx] = f2bf(hn);
                    if (fout) fout[idx] = hn;
                }
            }
            grid_barrier(counter, done += GRIDN);
        }

        // ---- phase 2: projection feedback (split-K across khalf) ----
        {
            const unsigned short* arow =
                hbuf + (size_t)(1 - p) * LBH + (size_t)(LL - 1) * BH + (size_t)m * HH;
            const unsigned short* bbase =
                pk_outw + (long)(n0 >> 4) * KT * 512 + lane * 16;
            const int kt0 = khalf * (KT / 2);
            const int kt1 = kt0 + KT / 2;

            v8f pacc = {};
            for (int kt = kt0; kt < kt1; ++kt) {
                ABpack a, b;
                load_a(a, arow, kt * 32, half);
                load_b(b, bbase + kt * 512);
                pacc = __builtin_amdgcn_wmma_f32_16x16x32_bf16(
                    false, a.v, false, b.v, (short)0, pacc, false, false);
            }
            if (khalf == 1) {
                #pragma unroll
                for (int r = 0; r < 8; ++r) red[tile][0][r][lane] = pacc[r];
            }
            __syncthreads();
            if (khalf == 0) {
                #pragma unroll
                for (int r = 0; r < 8; ++r) {
                    const int mrow = mstrip * 16 + half * 8 + r;
                    xbf[(size_t)mrow * HH + n] =
                        f2bf(pacc[r] + red[tile][0][r][lane] + obias);
                }
            }
            grid_barrier(counter, done += GRIDN);
        }
        p ^= 1;
    }
}

// ---------------------------------------------------------------------------
extern "C" void kernel_launch(void* const* d_in, const int* in_sizes, int n_in,
                              void* d_out, int out_size, void* d_ws, size_t ws_size,
                              hipStream_t stream) {
    const float* h0    = (const float*)d_in[0];
    const float* c0    = (const float*)d_in[1];
    const float* w_ih  = (const float*)d_in[2];   // [L,4H,H]
    const float* w_hh  = (const float*)d_in[3];   // [L,4H,H]
    const float* b_ih  = (const float*)d_in[4];   // [L,4H]
    const float* b_hh  = (const float*)d_in[5];   // [L,4H]
    const float* out_w = (const float*)d_in[6];   // [H,H]
    const float* out_b = (const float*)d_in[7];   // [H]
    float* out = (float*)d_out;                   // [T,B,H]

    // Workspace layout (base assumed >=256B aligned; all offsets 16B aligned)
    unsigned short* pk_wih  = (unsigned short*)d_ws;                        // L*4H*H bf16
    unsigned short* pk_whh  = pk_wih  + (size_t)LL * 4 * HH * HH;           // L*4H*H bf16
    unsigned short* pk_outw = pk_whh  + (size_t)LL * 4 * HH * HH;           // H*H bf16
    float*          bias    = (float*)(pk_outw + (size_t)HH * HH);          // L*4H f32
    float*          cst     = bias + (size_t)LL * 4 * HH;                   // L*B*H f32
    unsigned short* hbuf    = (unsigned short*)(cst + (size_t)LBH);         // 2*L*B*H bf16
    unsigned short* xbf     = hbuf + (size_t)2 * LBH;                       // B*H bf16
    unsigned*       counter = (unsigned*)(xbf + (size_t)BH);                // 1 u32

    // --- Pack weights into WMMA B-operand lane order (bf16) ---
    {
        const long tot_l = (long)LL * 4 * HH * HH;
        pack_w_kernel<<<(unsigned)((tot_l + 255) / 256), 256, 0, stream>>>(w_ih, pk_wih, HH, tot_l);
        pack_w_kernel<<<(unsigned)((tot_l + 255) / 256), 256, 0, stream>>>(w_hh, pk_whh, HH, tot_l);
        const long tot_o = (long)HH * HH;
        pack_w_kernel<<<(unsigned)((tot_o + 255) / 256), 256, 0, stream>>>(out_w, pk_outw, HH, tot_o);
    }
    // --- Init states, bias, x0, barrier counter (re-runs on every replay) ---
    lstm_init_kernel<<<(LBH + 255) / 256, 256, 0, stream>>>(
        h0, c0, b_ih, b_hh, hbuf, cst, bias, xbf, counter);

    // --- One persistent kernel runs the whole recurrence ---
    lstm_persistent<<<GRIDN, 512, 0, stream>>>(
        pk_wih, pk_whh, pk_outw, bias, out_b, cst, hbuf, xbf, out, counter);
}
